// Engram_14886356648848
// MI455X (gfx1250) — matrix operand order repaired and checked by reading, not compiled, not run on hardware
//
#include <hip/hip_runtime.h>
#include <hip/hip_bf16.h>

#define T_TOK   16384
#define HDIM    512
#define HCG     4
#define EDIM    768
#define NALL    2560   // 4*512 (keys) + 512 (v)

typedef __attribute__((ext_vector_type(16))) __bf16          v16bf;
typedef __attribute__((ext_vector_type(8)))  float           v8f;
typedef __attribute__((ext_vector_type(8)))  unsigned short  v8s;
typedef __attribute__((ext_vector_type(16))) unsigned short  v16s;

__device__ __forceinline__ unsigned short f2bf(float f) {
    unsigned int u = __float_as_uint(f);
    u += 0x7FFFu + ((u >> 16) & 1u);     // round-to-nearest-even
    return (unsigned short)(u >> 16);
}

__device__ __forceinline__ v16bf mkfrag(v8s lo, v8s hi) {
    v16s c = __builtin_shufflevector(lo, hi, 0,1,2,3,4,5,6,7,8,9,10,11,12,13,14,15);
    return __builtin_bit_cast(v16bf, c);
}

// CDNA5 async global->LDS copy (16 bytes per lane), tracked by ASYNCcnt.
// vDst = per-lane LDS byte offset (low 32 bits of the flat shared address).
__device__ __forceinline__ void async_b128(const unsigned short* g, const unsigned short* l) {
    unsigned loff = (unsigned)(size_t)l;
    asm volatile("global_load_async_to_lds_b128 %0, %1, off"
                 :: "v"(loff), "v"(g) : "memory");
}
#define WAIT_ASYNC_4 asm volatile("s_wait_asynccnt 0x4" ::: "memory")
#define WAIT_ASYNC_0 asm volatile("s_wait_asynccnt 0x0" ::: "memory")

// ---------------- Kernel 1: gather hash embeddings -> bf16 ----------------
// one thread = 4 floats; total T*12*16 threads
__global__ void k_gather(const int* __restrict__ hash_ids,
                         const int* __restrict__ offsets,
                         const float* __restrict__ embw,
                         unsigned short* __restrict__ embB) {
    int idx = blockIdx.x * 256 + threadIdx.x;       // < T*192
    int t = idx / 192;
    int r = idx - t * 192;
    int head = r >> 4, seg = r & 15;
    long row = (long)hash_ids[t * 12 + head] + (long)offsets[head];
    float4 e = *(const float4*)(embw + row * 64 + seg * 4);
    unsigned int lo = (unsigned int)f2bf(e.x) | ((unsigned int)f2bf(e.y) << 16);
    unsigned int hi = (unsigned int)f2bf(e.z) | ((unsigned int)f2bf(e.w) << 16);
    *(uint2*)(embB + (size_t)t * EDIM + head * 64 + seg * 4) = make_uint2(lo, hi);
}

// ---------------- Kernel 2: pack weights [keys(4) | vproj] -> bf16 --------
// W[k][n], k<768, n<2560
__global__ void k_wcvt(const float* __restrict__ kproj,
                       const float* __restrict__ vproj,
                       unsigned short* __restrict__ W) {
    int idx = blockIdx.x * 256 + threadIdx.x;       // < 768*2560
    int k = idx / NALL;
    int n = idx - k * NALL;
    float v;
    if (n < 2048) {
        int g = n >> 9, h = n & 511;
        v = kproj[(size_t)g * EDIM * HDIM + (size_t)k * HDIM + h];
    } else {
        v = vproj[(size_t)k * HDIM + (n - 2048)];
    }
    W[idx] = f2bf(v);
}

// ---------------- Kernel 3: WMMA GEMM  C[16384,2560] = A[16384,768] * B ----
// block tile 128x128, K-tile 32; 8 waves, each wave: 32(M) x 64(N)
// Double-buffered LDS fed by async global->LDS b128 copies (ASYNCcnt).
__global__ __launch_bounds__(256) void k_gemm(const unsigned short* __restrict__ A,
                                              const unsigned short* __restrict__ B,
                                              float* __restrict__ C) {
    __shared__ unsigned short As[2][128][40];   // 32 + 8 pad, row stride 80B (16B-mult)
    __shared__ unsigned short Bs[2][32][136];   // 128 + 8 pad, row stride 272B (16B-mult)

    const int tid  = threadIdx.x;
    const int lane = tid & 31;
    const int w    = tid >> 5;
    const int wm   = w & 3;          // 4 waves along M
    const int wn   = w >> 2;         // 2 waves along N
    const int m0   = blockIdx.y * 128;
    const int n0   = blockIdx.x * 128;

    v8f acc[2][4] = {};

    // global->LDS assignments (16 ushorts = 32 bytes per thread per matrix)
    const int arow = tid >> 1, aseg = tid & 1;   // A: 128 rows x 2 segs
    const int brow = tid >> 3, bseg = tid & 7;   // B: 32 rows x 8 segs
    const unsigned short* Ag = A + (size_t)(m0 + arow) * EDIM + aseg * 16;
    const unsigned short* Bg = B + (size_t)brow * NALL + n0 + bseg * 16;

    const int rr    = lane & 15;     // A fragment row within 16
    const int rhalf = lane >> 4;     // K-half selector

    auto issue_tile = [&](int kt, int buf) {
        const unsigned short* ga = Ag + (size_t)kt * 32;
        const unsigned short* gb = Bg + (size_t)kt * 32 * NALL;
        async_b128(ga,     &As[buf][arow][aseg * 16]);
        async_b128(ga + 8, &As[buf][arow][aseg * 16 + 8]);
        async_b128(gb,     &Bs[buf][brow][bseg * 16]);
        async_b128(gb + 8, &Bs[buf][brow][bseg * 16 + 8]);
    };

    issue_tile(0, 0);                            // prologue: tile 0 in flight

    for (int kt = 0; kt < EDIM / 32; ++kt) {
        const int buf = kt & 1;
        if (kt < EDIM / 32 - 1) {
            issue_tile(kt + 1, buf ^ 1);         // stream next tile during compute
            WAIT_ASYNC_4;                        // tile kt (first 4 ops) has landed
        } else {
            WAIT_ASYNC_0;
        }
        __syncthreads();                         // all waves' tile-kt data visible

        // A fragments: lane (half h) holds K {0..7,16..23} (h=0) / {8..15,24..31} (h=1)
        v16bf af[2];
#pragma unroll
        for (int mt = 0; mt < 2; ++mt) {
            const unsigned short* row = &As[buf][wm * 32 + mt * 16 + rr][0];
            v8s lo = *(const v8s*)(row + rhalf * 8);
            v8s hi = *(const v8s*)(row + 16 + rhalf * 8);
            af[mt] = mkfrag(lo, hi);
        }
        // B fragments: lane column N = lane&15, elements K {0..7,16..23} (h=0) / {8..15,24..31} (h=1)
        v16bf bfr[4];
#pragma unroll
        for (int nt = 0; nt < 4; ++nt) {
            const unsigned short* col = &Bs[buf][0][wn * 64 + nt * 16 + rr];
            v16s tb;
#pragma unroll
            for (int s = 0; s < 8; ++s) {
                tb[s]     = col[(rhalf * 8 + s) * 136];
                tb[s + 8] = col[(16 + rhalf * 8 + s) * 136];
            }
            bfr[nt] = __builtin_bit_cast(v16bf, tb);
        }
#pragma unroll
        for (int mt = 0; mt < 2; ++mt)
#pragma unroll
            for (int nt = 0; nt < 4; ++nt)
                acc[mt][nt] = __builtin_amdgcn_wmma_f32_16x16x32_bf16(
                    false, af[mt], false, bfr[nt], (short)0, acc[mt][nt], false, false);

        __syncthreads();                         // tile-kt reads done -> buf reusable
    }

    // epilogue: C/D layout: col = lane&15, row = vgpr + 8*(lane>>4)
    const int colh = lane & 15;
    const int rowh = (lane >> 4) * 8;
#pragma unroll
    for (int mt = 0; mt < 2; ++mt)
#pragma unroll
        for (int nt = 0; nt < 4; ++nt) {
            int col = n0 + wn * 64 + nt * 16 + colh;
            int rb  = m0 + wm * 32 + mt * 16 + rowh;
#pragma unroll
            for (int r = 0; r < 8; ++r)
                C[(size_t)(rb + r) * NALL + col] = acc[mt][nt][r];
        }
}

// ---------------- Kernel 4: rms norms, gate, value, x_n -------------------
__device__ __forceinline__ void reduce3(float& a, float& b, float& c, float (*s)[8]) {
#pragma unroll
    for (int o = 16; o > 0; o >>= 1) {
        a += __shfl_xor(a, o, 32);
        b += __shfl_xor(b, o, 32);
        c += __shfl_xor(c, o, 32);
    }
    int lane = threadIdx.x & 31, w = threadIdx.x >> 5;
    if (lane == 0) { s[0][w] = a; s[1][w] = b; s[2][w] = c; }
    __syncthreads();
    a = 0.f; b = 0.f; c = 0.f;
#pragma unroll
    for (int i = 0; i < 8; ++i) { a += s[0][i]; b += s[1][i]; c += s[2][i]; }
    __syncthreads();
}

__global__ __launch_bounds__(256) void k_gate(const float* __restrict__ keysv,
                                              const float* __restrict__ hidden,
                                              const float* __restrict__ qw,
                                              const float* __restrict__ kw,
                                              const float* __restrict__ cw,
                                              float* __restrict__ out,
                                              float* __restrict__ xn) {
    __shared__ float red[3][8];
    const int t = blockIdx.x;
    const int h = threadIdx.x * 2;                   // each thread: 2 elems
    const float* krow = keysv + (size_t)t * NALL;

    float2 v = *(const float2*)(krow + 2048 + h);    // vproj row, shared by groups
    float a = v.x * v.x + v.y * v.y, b = 0.f, c = 0.f;
    reduce3(a, b, c, red);
    const float sumv2 = a;

    for (int g = 0; g < 4; ++g) {
        float2 kv  = *(const float2*)(krow + g * 512 + h);
        float2 qv  = *(const float2*)(hidden + (size_t)t * 2048 + g * 512 + h);
        float2 kwv = *(const float2*)(kw + g * 512 + h);
        float2 qwv = *(const float2*)(qw + g * 512 + h);
        float kk = kv.x * kv.x + kv.y * kv.y;
        float qq = qv.x * qv.x + qv.y * qv.y;
        float kq = kv.x * qv.x * kwv.x * qwv.x + kv.y * qv.y * kwv.y * qwv.y;
        reduce3(kk, qq, kq, red);
        float invk = rsqrtf(kk * (1.0f / 512.0f) + 1e-5f);
        float invq = rsqrtf(qq * (1.0f / 512.0f) + 1e-5f);
        float s    = kq * invk * invq * 0.04419417382415922f;   // 1/sqrt(512)
        float gate = 1.0f / (1.0f + __expf(-s));
        float2 val = make_float2(gate * v.x, gate * v.y);
        *(float2*)(out + (size_t)t * 2048 + g * 512 + h) = val;
        float invv = rsqrtf(gate * gate * sumv2 * (1.0f / 512.0f) + 1e-5f);
        float2 cwv = *(const float2*)(cw + g * 512 + h);
        *(float2*)(xn + (size_t)t * 2048 + g * 512 + h) =
            make_float2(val.x * invv * cwv.x, val.y * invv * cwv.y);
    }
}

// ---------------- Kernel 5: dilated depthwise conv + SiLU + residual ------
__global__ void k_conv(const float* __restrict__ xn,
                       const float* __restrict__ wv,
                       float* __restrict__ out) {
    size_t idx = (size_t)blockIdx.x * 256 + threadIdx.x;   // < T*2048
    int t = (int)(idx >> 11);
    int c = (int)(idx & 2047);
    float4 w = *(const float4*)(wv + (size_t)c * 4);
    float acc = w.w * xn[idx];
    if (t >= 4)  acc += w.z * xn[idx - (size_t)4  * 2048];
    if (t >= 8)  acc += w.y * xn[idx - (size_t)8  * 2048];
    if (t >= 12) acc += w.x * xn[idx - (size_t)12 * 2048];
    out[idx] += acc / (1.0f + __expf(-acc));               // silu
}

extern "C" void kernel_launch(void* const* d_in, const int* in_sizes, int n_in,
                              void* d_out, int out_size, void* d_ws, size_t ws_size,
                              hipStream_t stream) {
    const int*   hash_ids = (const int*)d_in[0];
    const int*   offsets  = (const int*)d_in[1];
    const float* embw     = (const float*)d_in[2];
    const float* kproj    = (const float*)d_in[3];
    const float* qw       = (const float*)d_in[4];
    const float* kw       = (const float*)d_in[5];
    const float* vproj    = (const float*)d_in[6];
    const float* convw    = (const float*)d_in[7];
    const float* cnw      = (const float*)d_in[8];
    const float* hidden   = (const float*)d_in[9];
    float* out = (float*)d_out;

    char* ws = (char*)d_ws;
    unsigned short* embB  = (unsigned short*)ws;                 // T*768*2      = 25165824
    unsigned short* Wb    = (unsigned short*)(ws + 25165824);    // 768*2560*2   =  3932160
    float*          keysv = (float*)(ws + 29097984);             // T*2560*4     = 167772160
    float*          xn    = (float*)(ws + 196870144);            // T*2048*4     = 134217728

    k_gather<<<(T_TOK * 192) / 256, 256, 0, stream>>>(hash_ids, offsets, embw, embB);
    k_wcvt<<<(EDIM * NALL) / 256, 256, 0, stream>>>(kproj, vproj, Wb);
    dim3 grid(NALL / 128, T_TOK / 128);
    k_gemm<<<grid, 256, 0, stream>>>(embB, Wb, keysv);
    k_gate<<<T_TOK, 256, 0, stream>>>(keysv, hidden, qw, kw, cnw, out, xn);
    k_conv<<<(T_TOK * 2048) / 256, 256, 0, stream>>>(xn, convw, out);
}
